// FeatureFusionModule_20770461844126
// MI455X (gfx1250) — compile-verified, matrix operand order checked by reading, not run on hardware
//
#include <hip/hip_runtime.h>

typedef __bf16 bf16_t;
typedef __attribute__((ext_vector_type(16))) __bf16 v16bf;
typedef __attribute__((ext_vector_type(8)))  float  v8f;

union Frag { v16bf v; uint4 q[2]; };

#define LPITCH 40   // LDS row pitch in halves (80B, 16B aligned)

// CDNA5 async global->LDS path (ASYNCcnt-tracked DMA), with safe fallback.
// Probe-discovered signature: (int4 as1* src, int4 as3* dst, imm off, imm cpol)
#if defined(__gfx1250__) && __has_builtin(__builtin_amdgcn_global_load_async_to_lds_b128)
#define USE_ASYNC 1
typedef int vint4 __attribute__((vector_size(16)));
typedef __attribute__((address_space(1))) vint4 g1v4;
typedef __attribute__((address_space(3))) vint4 l3v4;
#else
#define USE_ASYNC 0
#endif

__device__ __forceinline__ void wait_async0() {
#if defined(__gfx1250__)
#if __has_builtin(__builtin_amdgcn_s_wait_asynccnt)
  __builtin_amdgcn_s_wait_asynccnt(0);
#else
  asm volatile("s_wait_asynccnt 0" ::: "memory");
#endif
#endif
}

// ---------------------------------------------------------------------------
// Generic tiled GEMM:  C[m,o] = sum_k A[m,k] * Bw[o,k]   (both K-contiguous)
// Block = 256 threads = 8 waves; tile 128(M) x 128(N), K-step 32.
// Wave w owns a 32x64 region: 2 M-tiles x 4 N-tiles of 16x16 accumulators.
// Async path: double-buffered LDS; tile i+1 DMA overlaps tile i WMMA.
// ---------------------------------------------------------------------------
template<bool RELU, bool SPLIT, bool OUTF32, bool ADDX>
__global__ __launch_bounds__(256) void gemm_wmma(
    const bf16_t* __restrict__ A, const bf16_t* __restrict__ Bw,
    const float* __restrict__ bias, const bf16_t* __restrict__ extra,
    bf16_t* __restrict__ outB, bf16_t* __restrict__ outB2, float* __restrict__ outF,
    int Mdim, int Ndim, int Kdim,
    int lda, int ldb, int ldc, int ld2, int split, int lde,
    int NH, long long sAb, long long sAh, long long sBb, long long sBh,
    long long sCb, long long sCh)
{
  const int z  = blockIdx.z;
  const int bb = z / NH, hh = z % NH;
  A  += (size_t)bb * sAb + (size_t)hh * sAh;
  Bw += (size_t)bb * sBb + (size_t)hh * sBh;
  const size_t coff = (size_t)bb * sCb + (size_t)hh * sCh;

  const int tid   = threadIdx.x;
  const int lane  = tid & 31;
  const int w     = tid >> 5;
  const int lrow  = lane & 15;
  const int lhalf = lane >> 4;
  const int koff  = lhalf * 8;
  const int m0 = blockIdx.y * 128, n0 = blockIdx.x * 128;
  const int m_off = (w >> 1) * 32, n_off = (w & 1) * 64;

  const int arow = tid >> 1;
  const int aoff = (tid & 1) * 16;

  v8f acc[2][4] = {};

#if USE_ASYNC
  __shared__ bf16_t As[2 * 128 * LPITCH];
  __shared__ bf16_t Bs[2 * 128 * LPITCH];

  // Pre-zero rows the async DMA never writes (OOB tile rows).
  {
    uint4 zz = {};
    if (m0 + arow >= Mdim) {
      uint4* d = (uint4*)&As[arow * LPITCH + aoff];
      d[0] = zz; d[1] = zz;
      uint4* d1 = (uint4*)&As[128 * LPITCH + arow * LPITCH + aoff];
      d1[0] = zz; d1[1] = zz;
    }
    if (n0 + arow >= Ndim) {
      uint4* d = (uint4*)&Bs[arow * LPITCH + aoff];
      d[0] = zz; d[1] = zz;
      uint4* d1 = (uint4*)&Bs[128 * LPITCH + arow * LPITCH + aoff];
      d1[0] = zz; d1[1] = zz;
    }
  }
  __syncthreads();

  auto issue_tile = [&](int k0, int buf) {
    const int gm = m0 + arow;
    if (gm < Mdim) {
      unsigned long long g =
          (unsigned long long)(A + (size_t)gm * lda + k0 + aoff);
      unsigned l = (unsigned)(unsigned long long)
          &As[buf * 128 * LPITCH + arow * LPITCH + aoff];
      __builtin_amdgcn_global_load_async_to_lds_b128((g1v4*)g, (l3v4*)(size_t)l, 0, 0);
      __builtin_amdgcn_global_load_async_to_lds_b128((g1v4*)g, (l3v4*)(size_t)l, 32, 0);
    }
    const int gn = n0 + arow;
    if (gn < Ndim) {
      unsigned long long g =
          (unsigned long long)(Bw + (size_t)gn * ldb + k0 + aoff);
      unsigned l = (unsigned)(unsigned long long)
          &Bs[buf * 128 * LPITCH + arow * LPITCH + aoff];
      __builtin_amdgcn_global_load_async_to_lds_b128((g1v4*)g, (l3v4*)(size_t)l, 0, 0);
      __builtin_amdgcn_global_load_async_to_lds_b128((g1v4*)g, (l3v4*)(size_t)l, 32, 0);
    }
  };

  const int T = Kdim >> 5;
  issue_tile(0, 0);
  for (int t = 0; t < T; ++t) {
    wait_async0();       // tile t is in LDS (this wave's DMAs)
    __syncthreads();     // everyone's tile-t writes visible; tile t-1 reads done
    if (t + 1 < T) issue_tile((t + 1) * 32, (t + 1) & 1);  // overlap with compute
    const bf16_t* Ab = &As[(t & 1) * 128 * LPITCH];
    const bf16_t* Bb = &Bs[(t & 1) * 128 * LPITCH];

    Frag af[2], bf[4];
#pragma unroll
    for (int mt = 0; mt < 2; ++mt) {
      const int base = (m_off + mt * 16 + lrow) * LPITCH;
      af[mt].q[0] = *(const uint4*)&Ab[base + koff];
      af[mt].q[1] = *(const uint4*)&Ab[base + 16 + koff];
    }
#pragma unroll
    for (int nt = 0; nt < 4; ++nt) {
      const int base = (n_off + nt * 16 + lrow) * LPITCH;
      bf[nt].q[0] = *(const uint4*)&Bb[base + koff];
      bf[nt].q[1] = *(const uint4*)&Bb[base + 16 + koff];
    }
#pragma unroll
    for (int mt = 0; mt < 2; ++mt)
#pragma unroll
      for (int nt = 0; nt < 4; ++nt)
        acc[mt][nt] = __builtin_amdgcn_wmma_f32_16x16x32_bf16(
            false, af[mt].v, false, bf[nt].v, (short)0, acc[mt][nt], false, false);
  }
#else
  __shared__ bf16_t As[128 * LPITCH];
  __shared__ bf16_t Bs[128 * LPITCH];

  for (int k0 = 0; k0 < Kdim; k0 += 32) {
    {
      const int gm = m0 + arow;
      uint4 q0 = {}, q1 = {};
      if (gm < Mdim) {
        const uint4* s = (const uint4*)(A + (size_t)gm * lda + k0 + aoff);
        q0 = s[0]; q1 = s[1];
        __builtin_prefetch(A + (size_t)gm * lda + k0 + 64, 0, 1);
      }
      uint4* d = (uint4*)&As[arow * LPITCH + aoff];
      d[0] = q0; d[1] = q1;
    }
    {
      const int gn = n0 + arow;
      uint4 q0 = {}, q1 = {};
      if (gn < Ndim) {
        const uint4* s = (const uint4*)(Bw + (size_t)gn * ldb + k0 + aoff);
        q0 = s[0]; q1 = s[1];
        __builtin_prefetch(Bw + (size_t)gn * ldb + k0 + 64, 0, 1);
      }
      uint4* d = (uint4*)&Bs[arow * LPITCH + aoff];
      d[0] = q0; d[1] = q1;
    }
    __syncthreads();

    Frag af[2], bf[4];
#pragma unroll
    for (int mt = 0; mt < 2; ++mt) {
      const int base = (m_off + mt * 16 + lrow) * LPITCH;
      af[mt].q[0] = *(const uint4*)&As[base + koff];
      af[mt].q[1] = *(const uint4*)&As[base + 16 + koff];
    }
#pragma unroll
    for (int nt = 0; nt < 4; ++nt) {
      const int base = (n_off + nt * 16 + lrow) * LPITCH;
      bf[nt].q[0] = *(const uint4*)&Bs[base + koff];
      bf[nt].q[1] = *(const uint4*)&Bs[base + 16 + koff];
    }
#pragma unroll
    for (int mt = 0; mt < 2; ++mt)
#pragma unroll
      for (int nt = 0; nt < 4; ++nt)
        acc[mt][nt] = __builtin_amdgcn_wmma_f32_16x16x32_bf16(
            false, af[mt].v, false, bf[nt].v, (short)0, acc[mt][nt], false, false);
    __syncthreads();
  }
#endif

  // Epilogue: D VGPR j -> row = j + lhalf*8, col = lrow within each 16x16 tile
#pragma unroll
  for (int mt = 0; mt < 2; ++mt) {
#pragma unroll
    for (int nt = 0; nt < 4; ++nt) {
      const int colb = n0 + n_off + nt * 16 + lrow;
      const int rowb = m0 + m_off + mt * 16 + lhalf * 8;
      if (colb >= Ndim) continue;
      const float bs = bias ? bias[colb] : 0.0f;
#pragma unroll
      for (int j = 0; j < 8; ++j) {
        const int row = rowb + j;
        if (row >= Mdim) continue;
        float v = acc[mt][nt][j] + bs;
        if (RELU) v = fmaxf(v, 0.0f);
        if (ADDX) v += (float)extra[(size_t)row * lde + colb];
        if (OUTF32) {
          outF[(size_t)row * ldc + colb + coff] = v;
        } else if (SPLIT) {
          if (colb < split) outB[(size_t)row * ldc + colb] = (bf16_t)v;
          else              outB2[(size_t)row * ld2 + (colb - split)] = (bf16_t)v;
        } else {
          outB[(size_t)row * ldc + colb + coff] = (bf16_t)v;
        }
      }
    }
  }
}

// ---------------------------------------------------------------------------
// Per-(b,h) context GEMM: ctx[d,e] = scale * sum_n k[n,d] * v[n,e]
// k = kv[:, h*64 .. +64), v = kv[:, 512 + h*64 .. +64); K = n = 4096.
// One block per (b,h); stages transposed 64x128 tiles of k,v in LDS.
// ---------------------------------------------------------------------------
__global__ __launch_bounds__(256) void ctx_wmma(
    const bf16_t* __restrict__ kv, float* __restrict__ ctx, float scale)
{
  const int z = blockIdx.x;
  const int b = z >> 3, h = z & 7;
  const bf16_t* base = kv + (size_t)b * 4096 * 1024;
  const int kcol = h * 64, vcol = 512 + h * 64;

  __shared__ bf16_t kT[64 * 136];
  __shared__ bf16_t vT[64 * 136];

  const int tid = threadIdx.x, lane = tid & 31, w = tid >> 5;
  const int lrow = lane & 15, lhalf = lane >> 4, koff = lhalf * 8;
  const int dt = w >> 1, et0 = (w & 1) * 2;

  v8f acc[2] = {};

  for (int n0 = 0; n0 < 4096; n0 += 128) {
    for (int i = tid; i < 128 * 64; i += 256) {
      const int r = i >> 6, c = i & 63;
      const size_t src = (size_t)(n0 + r) * 1024;
      kT[c * 136 + r] = base[src + kcol + c];
      vT[c * 136 + r] = base[src + vcol + c];
    }
    __syncthreads();
#pragma unroll
    for (int ks = 0; ks < 4; ++ks) {
      const int kk = ks * 32;
      Frag a, b0, b1;
      const int ab = (dt * 16 + lrow) * 136 + kk;
      a.q[0] = *(const uint4*)&kT[ab + koff];
      a.q[1] = *(const uint4*)&kT[ab + 16 + koff];
      const int e0 = (et0 * 16 + lrow) * 136 + kk;
      const int e1 = ((et0 + 1) * 16 + lrow) * 136 + kk;
      b0.q[0] = *(const uint4*)&vT[e0 + koff];
      b0.q[1] = *(const uint4*)&vT[e0 + 16 + koff];
      b1.q[0] = *(const uint4*)&vT[e1 + koff];
      b1.q[1] = *(const uint4*)&vT[e1 + 16 + koff];
      acc[0] = __builtin_amdgcn_wmma_f32_16x16x32_bf16(false, a.v, false, b0.v,
                                                       (short)0, acc[0], false, false);
      acc[1] = __builtin_amdgcn_wmma_f32_16x16x32_bf16(false, a.v, false, b1.v,
                                                       (short)0, acc[1], false, false);
    }
    __syncthreads();
  }
#pragma unroll
  for (int nt = 0; nt < 2; ++nt)
#pragma unroll
    for (int j = 0; j < 8; ++j) {
      const int d = dt * 16 + lhalf * 8 + j;
      const int e = (et0 + nt) * 16 + lrow;
      ctx[(size_t)z * 4096 + d * 64 + e] = acc[nt][j] * scale;
    }
}

// softmax over d (axis=-2) per (b,h,e); writes transposed bf16 ctxT[e][d]
__global__ __launch_bounds__(64) void softmax_k(
    const float* __restrict__ ctx, bf16_t* __restrict__ ctxT)
{
  const int z = blockIdx.x, e = threadIdx.x;
  const float* p = ctx + (size_t)z * 4096;
  float mx = -3.0e38f;
  for (int d = 0; d < 64; ++d) mx = fmaxf(mx, p[d * 64 + e]);
  float s = 0.0f;
  for (int d = 0; d < 64; ++d) s += __expf(p[d * 64 + e] - mx);
  const float inv = 1.0f / s;
  for (int d = 0; d < 64; ++d)
    ctxT[(size_t)z * 4096 + e * 64 + d] = (bf16_t)(__expf(p[d * 64 + e] - mx) * inv);
}

// (B,C,N) f32 -> (B,N,C) bf16 transpose via 32x32 LDS tile
__global__ __launch_bounds__(256) void transpose_to_bf16(
    const float* __restrict__ src, bf16_t* __restrict__ dst)
{
  __shared__ float tile[32][33];
  const int n0 = blockIdx.x * 32, c0 = blockIdx.y * 32, b = blockIdx.z;
  const int tx = threadIdx.x, ty = threadIdx.y;
#pragma unroll
  for (int i = 0; i < 4; ++i) {
    const int c = c0 + ty + i * 8;
    tile[ty + i * 8][tx] = src[((size_t)b * 512 + c) * 4096 + n0 + tx];
  }
  __syncthreads();
#pragma unroll
  for (int i = 0; i < 4; ++i) {
    const int n = n0 + ty + i * 8;
    dst[((size_t)b * 4096 + n) * 512 + c0 + tx] = (bf16_t)tile[tx][ty + i * 8];
  }
}

__global__ __launch_bounds__(256) void cvt_k(
    const float* __restrict__ s, bf16_t* __restrict__ d, int n)
{
  const int i = blockIdx.x * 256 + threadIdx.x;
  if (i < n) d[i] = (bf16_t)s[i];
}

// row LayerNorm over 512 f32, writes bf16 into fusion at column offset
__global__ __launch_bounds__(128) void layernorm_k(
    const float* __restrict__ in, const float* __restrict__ g,
    const float* __restrict__ bb, bf16_t* __restrict__ out, int colOff)
{
  const int row = blockIdx.x, tid = threadIdx.x;
  const float* p = in + (size_t)row * 512;
  float v[4], s = 0.0f, s2 = 0.0f;
#pragma unroll
  for (int i = 0; i < 4; ++i) {
    v[i] = p[tid + i * 128];
    s += v[i]; s2 += v[i] * v[i];
  }
  __shared__ float rs[128], rq[128];
  rs[tid] = s; rq[tid] = s2;
  __syncthreads();
  for (int st = 64; st > 0; st >>= 1) {
    if (tid < st) { rs[tid] += rs[tid + st]; rq[tid] += rq[tid + st]; }
    __syncthreads();
  }
  const float mean = rs[0] * (1.0f / 512.0f);
  const float var  = rq[0] * (1.0f / 512.0f) - mean * mean;
  const float rstd = rsqrtf(var + 1e-5f);
  bf16_t* o = out + (size_t)row * 1024 + colOff;
#pragma unroll
  for (int i = 0; i < 4; ++i) {
    const int c = tid + i * 128;
    o[c] = (bf16_t)((v[i] - mean) * rstd * g[c] + bb[c]);
  }
}

// NHWC depthwise 3x3 + bias + relu  (layout (b, h*64+w, c), C=512)
__global__ __launch_bounds__(256) void dwconv_k(
    const bf16_t* __restrict__ t1, const float* __restrict__ wdw,
    const float* __restrict__ bdw, bf16_t* __restrict__ t2)
{
  const size_t idx = (size_t)blockIdx.x * 256 + threadIdx.x;
  const int c  = (int)(idx & 511);
  const int wq = (int)((idx >> 9) & 63);
  const int hq = (int)((idx >> 15) & 63);
  const int b  = (int)(idx >> 21);
  float acc = bdw[c];
#pragma unroll
  for (int ky = 0; ky < 3; ++ky) {
    const int hh = hq + ky - 1;
    if (hh < 0 || hh > 63) continue;
#pragma unroll
    for (int kx = 0; kx < 3; ++kx) {
      const int ww = wq + kx - 1;
      if (ww < 0 || ww > 63) continue;
      const size_t src = (((size_t)b * 4096 + hh * 64 + ww) << 9) + c;
      acc += wdw[c * 9 + ky * 3 + kx] * (float)t1[src];
    }
  }
  t2[idx] = (bf16_t)fmaxf(acc, 0.0f);
}

// per-channel BN stats. phase 0: stats of t3. phase 1: stats of
// residual + (t3 - mu1)*rstd1*g1 + b1
__global__ __launch_bounds__(256) void bn_stats_k(
    const float* __restrict__ t3, const float* __restrict__ residual,
    const float* __restrict__ mu1, const float* __restrict__ rstd1,
    const float* __restrict__ g1, const float* __restrict__ b1,
    float* __restrict__ muO, float* __restrict__ rstdO, int phase)
{
  const int c = blockIdx.x, tid = threadIdx.x;
  float s = 0.0f, s2 = 0.0f;
  float m1 = 0.0f, r1 = 1.0f, gg = 1.0f, bb = 0.0f;
  if (phase) { m1 = mu1[c]; r1 = rstd1[c]; gg = g1[c]; bb = b1[c]; }
  for (int m = tid; m < 32768; m += 256) {
    float v = t3[(size_t)m * 512 + c];
    if (phase) v = residual[(size_t)m * 512 + c] + (v - m1) * r1 * gg + bb;
    s += v; s2 += v * v;
  }
  __shared__ float rs[256], rq[256];
  rs[tid] = s; rq[tid] = s2;
  __syncthreads();
  for (int st = 128; st > 0; st >>= 1) {
    if (tid < st) { rs[tid] += rs[tid + st]; rq[tid] += rq[tid + st]; }
    __syncthreads();
  }
  if (tid == 0) {
    const float mean = rs[0] * (1.0f / 32768.0f);
    const float var  = rq[0] * (1.0f / 32768.0f) - mean * mean;
    muO[c] = mean;
    rstdO[c] = rsqrtf(var + 1e-5f);
  }
}

// final: out_nchw = bn2( residual + bn1(t3) ); one block per (b,c)
__global__ __launch_bounds__(256) void bn_apply_k(
    const float* __restrict__ t3, const float* __restrict__ residual,
    const float* __restrict__ mu1, const float* __restrict__ rstd1,
    const float* __restrict__ g1, const float* __restrict__ b1,
    const float* __restrict__ mu2, const float* __restrict__ rstd2,
    const float* __restrict__ g2, const float* __restrict__ b2,
    float* __restrict__ out)
{
  const int bc = blockIdx.x;
  const int b = bc >> 9, c = bc & 511;
  const float m1 = mu1[c], r1 = rstd1[c], gg1 = g1[c], bb1 = b1[c];
  const float m2 = mu2[c], r2 = rstd2[c], gg2 = g2[c], bb2 = b2[c];
  for (int hw = threadIdx.x; hw < 4096; hw += 256) {
    const size_t m = (size_t)b * 4096 + hw;
    const float v = residual[m * 512 + c] + (t3[m * 512 + c] - m1) * r1 * gg1 + bb1;
    out[(size_t)bc * 4096 + hw] = (v - m2) * r2 * gg2 + bb2;
  }
}

// ---------------------------------------------------------------------------
extern "C" void kernel_launch(void* const* d_in, const int* in_sizes, int n_in,
                              void* d_out, int out_size, void* d_ws, size_t ws_size,
                              hipStream_t stream) {
  (void)in_sizes; (void)n_in; (void)out_size; (void)ws_size;
  const float* rgb        = (const float*)d_in[0];
  const float* x          = (const float*)d_in[1];
  const float* W_rgb_proj = (const float*)d_in[2];
  const float* b_rgb_proj = (const float*)d_in[3];
  const float* W_x_proj   = (const float*)d_in[4];
  const float* b_x_proj   = (const float*)d_in[5];
  const float* W_kv_rgb   = (const float*)d_in[6];
  const float* W_kv_x     = (const float*)d_in[7];
  const float* W_out_rgb  = (const float*)d_in[8];
  const float* b_out_rgb  = (const float*)d_in[9];
  const float* W_out_x    = (const float*)d_in[10];
  const float* b_out_x    = (const float*)d_in[11];
  const float* ln_rgb_g   = (const float*)d_in[12];
  const float* ln_rgb_b   = (const float*)d_in[13];
  const float* ln_x_g     = (const float*)d_in[14];
  const float* ln_x_b     = (const float*)d_in[15];
  const float* W_res      = (const float*)d_in[16];
  const float* W_ce1      = (const float*)d_in[17];
  const float* b_ce1      = (const float*)d_in[18];
  const float* W_dw       = (const float*)d_in[19];
  const float* b_dw       = (const float*)d_in[20];
  const float* W_ce2      = (const float*)d_in[21];
  const float* b_ce2      = (const float*)d_in[22];
  const float* bn1_g      = (const float*)d_in[23];
  const float* bn1_b      = (const float*)d_in[24];
  const float* bn2_g      = (const float*)d_in[25];
  const float* bn2_b      = (const float*)d_in[26];
  float* out = (float*)d_out;

  const size_t M = 32768;           // B * H * W
  char* ws = (char*)d_ws;
  size_t off = 0;
  auto take = [&](size_t bytes) -> char* {
    off = (off + 255) & ~(size_t)255;
    char* p = ws + off;
    off += bytes;
    return p;
  };

  // bf16 weights
  bf16_t* wpr  = (bf16_t*)take(1024 * 512 * 2);
  bf16_t* wpx  = (bf16_t*)take(1024 * 512 * 2);
  bf16_t* wkvr = (bf16_t*)take(1024 * 512 * 2);
  bf16_t* wkvx = (bf16_t*)take(1024 * 512 * 2);
  bf16_t* wor  = (bf16_t*)take(512 * 1024 * 2);
  bf16_t* wox  = (bf16_t*)take(512 * 1024 * 2);
  bf16_t* wre  = (bf16_t*)take(512 * 1024 * 2);
  bf16_t* wc1  = (bf16_t*)take(512 * 1024 * 2);
  bf16_t* wc2  = (bf16_t*)take(512 * 512 * 2);
  // stats
  float* mu1   = (float*)take(512 * 4);
  float* rstd1 = (float*)take(512 * 4);
  float* mu2   = (float*)take(512 * 4);
  float* rstd2 = (float*)take(512 * 4);
  // activations
  bf16_t* r_bf = (bf16_t*)take(M * 512 * 2);
  bf16_t* x_bf = (bf16_t*)take(M * 512 * 2);
  bf16_t* u_r  = (bf16_t*)take(M * 512 * 2);
  bf16_t* u_x  = (bf16_t*)take(M * 512 * 2);
  bf16_t* yr   = (bf16_t*)take(M * 1024 * 2);
  bf16_t* yx   = (bf16_t*)take(M * 1024 * 2);
  bf16_t* kvr  = (bf16_t*)take(M * 1024 * 2);
  bf16_t* kvx  = (bf16_t*)take(M * 1024 * 2);
  float*  ctxr = (float*)take(64 * 4096 * 4);
  float*  ctxx = (float*)take(64 * 4096 * 4);
  bf16_t* ctxTr = (bf16_t*)take(64 * 4096 * 2);
  bf16_t* ctxTx = (bf16_t*)take(64 * 4096 * 2);
  // aliases (lifetimes disjoint)
  float*  ln_in    = (float*)kvr;     // after ctx done
  bf16_t* fusion   = (bf16_t*)kvx;    // after ctx done
  float*  residual = (float*)yr;      // after out-proj done
  float*  t3       = (float*)yx;      // after out-proj done
  bf16_t* t1       = u_r;             // after attention done
  bf16_t* t2       = u_x;

  const dim3 blk(256);
  auto cvgrid = [](int n) { return dim3((n + 255) / 256); };

  // 1) weights -> bf16
  cvt_k<<<cvgrid(1024 * 512), blk, 0, stream>>>(W_rgb_proj, wpr, 1024 * 512);
  cvt_k<<<cvgrid(1024 * 512), blk, 0, stream>>>(W_x_proj,   wpx, 1024 * 512);
  cvt_k<<<cvgrid(1024 * 512), blk, 0, stream>>>(W_kv_rgb,   wkvr, 1024 * 512);
  cvt_k<<<cvgrid(1024 * 512), blk, 0, stream>>>(W_kv_x,     wkvx, 1024 * 512);
  cvt_k<<<cvgrid(512 * 1024), blk, 0, stream>>>(W_out_rgb,  wor, 512 * 1024);
  cvt_k<<<cvgrid(512 * 1024), blk, 0, stream>>>(W_out_x,    wox, 512 * 1024);
  cvt_k<<<cvgrid(512 * 1024), blk, 0, stream>>>(W_res,      wre, 512 * 1024);
  cvt_k<<<cvgrid(512 * 1024), blk, 0, stream>>>(W_ce1,      wc1, 512 * 1024);
  cvt_k<<<cvgrid(512 * 512),  blk, 0, stream>>>(W_ce2,      wc2, 512 * 512);

  // 2) NCHW -> (B,N,C) bf16
  transpose_to_bf16<<<dim3(128, 16, 8), dim3(32, 8), 0, stream>>>(rgb, r_bf);
  transpose_to_bf16<<<dim3(128, 16, 8), dim3(32, 8), 0, stream>>>(x,   x_bf);

  const dim3 gN1024(8, 256, 1), gN512(4, 256, 1);

  // 3) input projections: relu(A@Wp^T + b) split into [y | u]
  gemm_wmma<true, true, false, false><<<gN1024, blk, 0, stream>>>(
      r_bf, wpr, b_rgb_proj, nullptr, yr, u_r, nullptr,
      32768, 1024, 512, 512, 512, 1024, 512, 512, 0, 1, 0, 0, 0, 0, 0, 0);
  gemm_wmma<true, true, false, false><<<gN1024, blk, 0, stream>>>(
      x_bf, wpx, b_x_proj, nullptr, yx, u_x, nullptr,
      32768, 1024, 512, 512, 512, 1024, 512, 512, 0, 1, 0, 0, 0, 0, 0, 0);

  // 4) KV projections
  gemm_wmma<false, false, false, false><<<gN1024, blk, 0, stream>>>(
      u_r, wkvr, nullptr, nullptr, kvr, nullptr, nullptr,
      32768, 1024, 512, 512, 512, 1024, 0, 0, 0, 1, 0, 0, 0, 0, 0, 0);
  gemm_wmma<false, false, false, false><<<gN1024, blk, 0, stream>>>(
      u_x, wkvx, nullptr, nullptr, kvx, nullptr, nullptr,
      32768, 1024, 512, 512, 512, 1024, 0, 0, 0, 1, 0, 0, 0, 0, 0, 0);

  // 5) per-head context (k^T v) * scale, then softmax over d -> ctxT bf16
  const float scale = 0.125f;  // (C/nh)^-0.5 = 64^-0.5
  ctx_wmma<<<dim3(64), blk, 0, stream>>>(kvr, ctxr, scale);
  ctx_wmma<<<dim3(64), blk, 0, stream>>>(kvx, ctxx, scale);
  softmax_k<<<dim3(64), dim3(64), 0, stream>>>(ctxr, ctxTr);
  softmax_k<<<dim3(64), dim3(64), 0, stream>>>(ctxx, ctxTx);

  // 6) o = q @ ctx (cross): per-(b,h) batched GEMM, writes cols 512.. of y*
  const dim3 g6(1, 32, 64);
  gemm_wmma<false, false, false, false><<<g6, blk, 0, stream>>>(
      u_r, ctxTx, nullptr, nullptr, yr + 512, nullptr, nullptr,
      4096, 64, 64, 512, 64, 1024, 0, 0, 0,
      8, (long long)4096 * 512, 64, (long long)8 * 4096, 4096,
      (long long)4096 * 1024, 64);
  gemm_wmma<false, false, false, false><<<g6, blk, 0, stream>>>(
      u_x, ctxTr, nullptr, nullptr, yx + 512, nullptr, nullptr,
      4096, 64, 64, 512, 64, 1024, 0, 0, 0,
      8, (long long)4096 * 512, 64, (long long)8 * 4096, 4096,
      (long long)4096 * 1024, 64);

  // 7) out projections + residual add, then LayerNorm -> fusion bf16
  gemm_wmma<false, false, true, true><<<gN512, blk, 0, stream>>>(
      yr, wor, b_out_rgb, r_bf, nullptr, nullptr, ln_in,
      32768, 512, 1024, 1024, 1024, 512, 0, 0, 512, 1, 0, 0, 0, 0, 0, 0);
  layernorm_k<<<dim3(32768), dim3(128), 0, stream>>>(ln_in, ln_rgb_g, ln_rgb_b, fusion, 0);
  gemm_wmma<false, false, true, true><<<gN512, blk, 0, stream>>>(
      yx, wox, b_out_x, x_bf, nullptr, nullptr, ln_in,
      32768, 512, 1024, 1024, 1024, 512, 0, 0, 512, 1, 0, 0, 0, 0, 0, 0);
  layernorm_k<<<dim3(32768), dim3(128), 0, stream>>>(ln_in, ln_x_g, ln_x_b, fusion, 512);

  // 8) residual path and channel-excitation path
  gemm_wmma<false, false, true, false><<<gN512, blk, 0, stream>>>(
      fusion, wre, nullptr, nullptr, nullptr, nullptr, residual,
      32768, 512, 1024, 1024, 1024, 512, 0, 0, 0, 1, 0, 0, 0, 0, 0, 0);
  gemm_wmma<false, false, false, false><<<gN512, blk, 0, stream>>>(
      fusion, wc1, b_ce1, nullptr, t1, nullptr, nullptr,
      32768, 512, 1024, 1024, 1024, 512, 0, 0, 0, 1, 0, 0, 0, 0, 0, 0);
  dwconv_k<<<dim3(65536), blk, 0, stream>>>(t1, W_dw, b_dw, t2);
  gemm_wmma<false, false, true, false><<<gN512, blk, 0, stream>>>(
      t2, wc2, b_ce2, nullptr, nullptr, nullptr, t3,
      32768, 512, 512, 512, 512, 512, 0, 0, 0, 1, 0, 0, 0, 0, 0, 0);

  // 9) batch norms and final NCHW output
  bn_stats_k<<<dim3(512), blk, 0, stream>>>(t3, residual, mu1, rstd1, bn1_g, bn1_b,
                                            mu1, rstd1, 0);
  bn_stats_k<<<dim3(512), blk, 0, stream>>>(t3, residual, mu1, rstd1, bn1_g, bn1_b,
                                            mu2, rstd2, 1);
  bn_apply_k<<<dim3(4096), blk, 0, stream>>>(t3, residual, mu1, rstd1, bn1_g, bn1_b,
                                             mu2, rstd2, bn2_g, bn2_b, out);
}
